// DyResA_AlexNet_15822659518913
// MI455X (gfx1250) — matrix operand order repaired and checked by reading, not compiled, not run on hardware
//
#include <hip/hip_runtime.h>
#include <hip/hip_bf16.h>

// ---------------------------------------------------------------------------
// Types for CDNA5 WMMA (wave32): D(16x16,f32) = A(16x32,bf16) x B(32x16,bf16)
// ---------------------------------------------------------------------------
typedef __attribute__((ext_vector_type(16))) __bf16 v16bf;
typedef __attribute__((ext_vector_type(8)))  float  v8f;

union Frag {
    v16bf v;
    uint4 u[2];
};

__device__ __forceinline__ __bf16 f2bf(float f) {
    unsigned u = __builtin_bit_cast(unsigned, f);
    unsigned r = (u + 0x7FFFu + ((u >> 16) & 1u)) >> 16;
    unsigned short s = (unsigned short)r;
    return __builtin_bit_cast(__bf16, s);
}

// ---------------------------------------------------------------------------
// Routing: mode 0 => softmax((avg+std) @ rw^T + rb)   (DyRes 'A')
//          mode 1 => sigmoid(avg @ rw^T + rb)         (CondConv)
// One block per sample. E = 3.
// ---------------------------------------------------------------------------
__global__ void routing_kernel(const float* __restrict__ act, int C, int HW,
                               const float* __restrict__ rw,
                               const float* __restrict__ rb,
                               int mode, float* __restrict__ rout) {
    const int b = blockIdx.x;
    __shared__ float part[3][256];
    float l0 = 0.f, l1 = 0.f, l2 = 0.f;
    for (int c = threadIdx.x; c < C; c += blockDim.x) {
        const float* p = act + ((size_t)b * C + c) * HW;
        float s1 = 0.f, s2 = 0.f;
        for (int i = 0; i < HW; ++i) { float v = p[i]; s1 += v; s2 += v * v; }
        float mean = s1 / (float)HW;
        float d = mean;
        if (mode == 0) {
            float var = s2 / (float)HW - mean * mean;
            d += sqrtf(fmaxf(var, 0.f));
        }
        l0 += d * rw[0 * C + c];
        l1 += d * rw[1 * C + c];
        l2 += d * rw[2 * C + c];
    }
    part[0][threadIdx.x] = l0; part[1][threadIdx.x] = l1; part[2][threadIdx.x] = l2;
    __syncthreads();
    for (int s = blockDim.x >> 1; s > 0; s >>= 1) {
        if ((int)threadIdx.x < s)
            for (int e = 0; e < 3; ++e)
                part[e][threadIdx.x] += part[e][threadIdx.x + s];
        __syncthreads();
    }
    if (threadIdx.x == 0) {
        float v0 = part[0][0] + rb[0];
        float v1 = part[1][0] + rb[1];
        float v2 = part[2][0] + rb[2];
        if (mode == 0) {
            float mx = fmaxf(v0, fmaxf(v1, v2));
            float e0 = expf(v0 - mx), e1 = expf(v1 - mx), e2 = expf(v2 - mx);
            float s = e0 + e1 + e2;
            rout[b * 3 + 0] = e0 / s; rout[b * 3 + 1] = e1 / s; rout[b * 3 + 2] = e2 / s;
        } else {
            rout[b * 3 + 0] = 1.f / (1.f + expf(-v0));
            rout[b * 3 + 1] = 1.f / (1.f + expf(-v1));
            rout[b * 3 + 2] = 1.f / (1.f + expf(-v2));
        }
    }
}

// ---------------------------------------------------------------------------
// Expert-mixed conv as per-sample im2col GEMM through WMMA.
// WAVES waves per block: each wave owns one Co tile (own A tile in LDS),
// all waves share NT im2col B-tiles staged cooperatively -> 1/WAVES the
// gather cost per WMMA. All dims compile-time; loops fully unroll.
// grid = (CO/(16*WAVES), n_tiles/NT, B); block = 32*WAVES.
// ---------------------------------------------------------------------------
template <int CI, int H, int W, int CO, int HO, int WO, int STRIDE, int KTOT,
          int NT, int WAVES>
__global__ __launch_bounds__(32 * WAVES)
void mixed_conv_wmma(const float* __restrict__ x, const float* __restrict__ experts,
                     const float* __restrict__ r, float* __restrict__ out) {
    constexpr int PAD  = 1;
    constexpr int NTOT = HO * WO;
    constexpr int EPT  = (NT * 512) / (32 * WAVES);   // B elems staged per thread
    static_assert((NT * 512) % (32 * WAVES) == 0 && EPT >= 1 && (32 % EPT) == 0, "");

    const int tid  = threadIdx.x;
    const int lane = tid & 31;
    const int wave = tid >> 5;
    const int b    = blockIdx.z;
    const int co0  = blockIdx.x * (16 * WAVES) + wave * 16;
    const int t0   = blockIdx.y * NT;

    const float r0 = r[b * 3 + 0], r1 = r[b * 3 + 1], r2 = r[b * 3 + 2];

    __shared__ __align__(16) __bf16 Al[WAVES][16 * 32];
    __shared__ __align__(16) __bf16 Bl[NT][16 * 32];

    v8f acc[NT] = {};

    // ---- A staging coords (per wave) ----
    const int rowm  = lane >> 1;
    const int kbase = (lane & 1) * 16;
    const float* w0 = experts + (size_t)(co0 + rowm) * (CI * 9);
    const float* w1 = w0 + (size_t)CO * CI * 9;
    const float* w2 = w1 + (size_t)CO * CI * 9;
    const float* xb = x + (size_t)b * CI * H * W;

    // ---- B staging coords (per thread, whole block cooperates) ----
    const int e0 = tid * EPT;          // linear index into [NT][16 n][32 k]
    const int bt = e0 >> 9;            // B tile
    const int bn = (e0 >> 5) & 15;     // n row within tile
    const int bk = e0 & 31;            // k start (contiguous run of EPT)
    const int ng = (t0 + bt) * 16 + bn;
    const int boh = ng / WO, bow = ng - boh * WO;

    for (int k0 = 0; k0 < KTOT; k0 += 32) {
        // ---- stage A: mix 3 expert rows (this wave's Co tile) ----
        if constexpr ((KTOT & 31) == 0) {
            #pragma unroll
            for (int q = 0; q < 4; ++q) {
                const int kq = k0 + kbase + q * 4;
                const float4 a = *(const float4*)(w0 + kq);
                const float4 c = *(const float4*)(w1 + kq);
                const float4 d = *(const float4*)(w2 + kq);
                __bf16* dst = &Al[wave][rowm * 32 + kbase + q * 4];
                dst[0] = f2bf(r0 * a.x + r1 * c.x + r2 * d.x);
                dst[1] = f2bf(r0 * a.y + r1 * c.y + r2 * d.y);
                dst[2] = f2bf(r0 * a.z + r1 * c.z + r2 * d.z);
                dst[3] = f2bf(r0 * a.w + r1 * c.w + r2 * d.w);
            }
        } else {
            #pragma unroll
            for (int i = 0; i < 16; ++i) {
                const int k = k0 + kbase + i;
                float v = 0.f;
                if (k < KTOT) v = r0 * w0[k] + r1 * w1[k] + r2 * w2[k];
                Al[wave][rowm * 32 + kbase + i] = f2bf(v);
            }
        }
        // ---- stage B: im2col, transposed [n][k], incremental (ci,kh,kw) walk ----
        {
            const int kk = k0 + bk;
            int ci = kk / 9;
            int rem = kk - ci * 9;
            int kh = rem / 3;
            int kw = rem - kh * 3;
            #pragma unroll
            for (int j = 0; j < EPT; ++j) {
                float v = 0.f;
                const bool kvalid = ((KTOT & 31) == 0) || (kk + j < KTOT);
                if (kvalid) {
                    const int ih = boh * STRIDE - PAD + kh;
                    const int iw = bow * STRIDE - PAD + kw;
                    if (ih >= 0 && ih < H && iw >= 0 && iw < W)
                        v = xb[(ci * H + ih) * W + iw];
                }
                Bl[bt][bn * 32 + bk + j] = f2bf(v);
                if (++kw == 3) { kw = 0; if (++kh == 3) { kh = 0; ++ci; } }
            }
        }
        __syncthreads();

        // ---- fragments + NT unrolled WMMAs (per wave) ----
        const int m = lane & 15, h = lane >> 4;
        Frag fa;
        fa.u[0] = *(const uint4*)&Al[wave][m * 32 + h * 8];       // K = h*8..+7
        fa.u[1] = *(const uint4*)&Al[wave][m * 32 + 16 + h * 8];  // K = 16+h*8..
        #pragma unroll
        for (int t = 0; t < NT; ++t) {
            Frag fb;
            fb.u[0] = *(const uint4*)&Bl[t][m * 32 + h * 8];
            fb.u[1] = *(const uint4*)&Bl[t][m * 32 + 16 + h * 8];
            acc[t] = __builtin_amdgcn_wmma_f32_16x16x32_bf16(
                false, fa.v, false, fb.v, (short)0, acc[t], false, false);
        }
        __syncthreads();
    }

    // ---- store D with fused ReLU ----
    const int coln = lane & 15, half = lane >> 4;
    #pragma unroll
    for (int t = 0; t < NT; ++t) {
        const int n = (t0 + t) * 16 + coln;
        #pragma unroll
        for (int rr = 0; rr < 8; ++rr) {
            const int row = rr + 8 * half;
            out[((size_t)b * CO + co0 + row) * NTOT + n] = fmaxf(acc[t][rr], 0.f);
        }
    }
}

// ---------------------------------------------------------------------------
// FC GEMM:  C[M,N] = A[M,K](bf16) * W[N,K]^T(bf16) + bias, optional ReLU.
// Fragments straight from global (two b128 loads each). M2 m-tiles per wave
// reuse every B fragment M2 times (halves W traffic). NT*M2 WMMAs per k-step.
// One wave per block; grid = (Ntiles/NT, M/(16*M2)); tbase offsets tiles.
// ---------------------------------------------------------------------------
template <int NT, int M2>
__global__ __launch_bounds__(32)
void fc_wmma(const __bf16* __restrict__ A, const __bf16* __restrict__ Wt,
             const float* __restrict__ bias, __bf16* __restrict__ outb,
             float* __restrict__ outf, int K, int N, int tbase, int do_relu) {
    const int lane = threadIdx.x;
    const int m0 = blockIdx.y * (16 * M2);
    const int t0 = tbase + blockIdx.x * NT;

    const int mrow = lane & 15, h = lane >> 4;

    const __bf16* arow[M2];
    #pragma unroll
    for (int mm = 0; mm < M2; ++mm)
        arow[mm] = A + (size_t)(m0 + mm * 16 + mrow) * K;

    const __bf16* wrow[NT];
    #pragma unroll
    for (int t = 0; t < NT; ++t) {
        int n = (t0 + t) * 16 + mrow;
        if (n >= N) n = N - 1;   // clamp for ragged last tile
        wrow[t] = Wt + (size_t)n * K;
    }

    v8f acc[M2][NT] = {};
    #pragma unroll 2
    for (int k0 = 0; k0 < K; k0 += 32) {
        Frag fa[M2];
        #pragma unroll
        for (int mm = 0; mm < M2; ++mm) {
            __builtin_prefetch(arow[mm] + k0 + 256, 0, 0);
            fa[mm].u[0] = *(const uint4*)&arow[mm][k0 + h * 8];
            fa[mm].u[1] = *(const uint4*)&arow[mm][k0 + 16 + h * 8];
        }
        #pragma unroll
        for (int t = 0; t < NT; ++t) {
            Frag fb;
            fb.u[0] = *(const uint4*)&wrow[t][k0 + h * 8];
            fb.u[1] = *(const uint4*)&wrow[t][k0 + 16 + h * 8];
            #pragma unroll
            for (int mm = 0; mm < M2; ++mm)
                acc[mm][t] = __builtin_amdgcn_wmma_f32_16x16x32_bf16(
                    false, fa[mm].v, false, fb.v, (short)0, acc[mm][t], false, false);
        }
    }

    const int coln = lane & 15, half = lane >> 4;
    #pragma unroll
    for (int t = 0; t < NT; ++t) {
        const int n = (t0 + t) * 16 + coln;
        if (n >= N) continue;
        const float bv = bias ? bias[n] : 0.f;
        #pragma unroll
        for (int mm = 0; mm < M2; ++mm) {
            #pragma unroll
            for (int rr = 0; rr < 8; ++rr) {
                const int row = m0 + mm * 16 + rr + 8 * half;
                float v = acc[mm][t][rr] + bv;
                if (do_relu) v = fmaxf(v, 0.f);
                if (outf) outf[(size_t)row * N + n] = v;
                if (outb) outb[(size_t)row * N + n] = f2bf(v);
            }
        }
    }
}

// ---------------------------------------------------------------------------
// 2x2 max-pool (f32 -> f32), and pool+flatten (f32 -> bf16) for the FC input.
// ---------------------------------------------------------------------------
__global__ void maxpool2_kernel(const float* __restrict__ in, float* __restrict__ out,
                                int C, int H, int W, long total) {
    const int Ho = H >> 1, Wo = W >> 1;
    for (long i = (long)blockIdx.x * blockDim.x + threadIdx.x; i < total;
         i += (long)gridDim.x * blockDim.x) {
        long t = i;
        int ow = (int)(t % Wo); t /= Wo;
        int oh = (int)(t % Ho); t /= Ho;
        int c  = (int)(t % C);  long b = t / C;
        const float* p = in + (((size_t)b * C + c) * H + oh * 2) * W + ow * 2;
        out[i] = fmaxf(fmaxf(p[0], p[1]), fmaxf(p[W], p[W + 1]));
    }
}

__global__ void maxpool2_bf16_kernel(const float* __restrict__ in, __bf16* __restrict__ out,
                                     int C, int H, int W, long total) {
    const int Ho = H >> 1, Wo = W >> 1;
    for (long i = (long)blockIdx.x * blockDim.x + threadIdx.x; i < total;
         i += (long)gridDim.x * blockDim.x) {
        long t = i;
        int ow = (int)(t % Wo); t /= Wo;
        int oh = (int)(t % Ho); t /= Ho;
        int c  = (int)(t % C);  long b = t / C;
        const float* p = in + (((size_t)b * C + c) * H + oh * 2) * W + ow * 2;
        out[i] = f2bf(fmaxf(fmaxf(p[0], p[1]), fmaxf(p[W], p[W + 1])));
    }
}

__global__ void f32_to_bf16_kernel(const float* __restrict__ in, __bf16* __restrict__ out, long n) {
    for (long i = (long)blockIdx.x * blockDim.x + threadIdx.x; i < n;
         i += (long)gridDim.x * blockDim.x)
        out[i] = f2bf(in[i]);
}

// ---------------------------------------------------------------------------
// Launch
// ---------------------------------------------------------------------------
extern "C" void kernel_launch(void* const* d_in, const int* in_sizes, int n_in,
                              void* d_out, int out_size, void* d_ws, size_t ws_size,
                              hipStream_t stream) {
    const int B = 512;
    const float* x    = (const float*)d_in[0];
    const float* dw1  = (const float*)d_in[1];
    const float* rw1  = (const float*)d_in[2];
    const float* rb1  = (const float*)d_in[3];
    const float* dw2  = (const float*)d_in[4];
    const float* rw2  = (const float*)d_in[5];
    const float* rb2  = (const float*)d_in[6];
    const float* dw3  = (const float*)d_in[7];
    const float* rw3  = (const float*)d_in[8];
    const float* rb3  = (const float*)d_in[9];
    const float* cw4  = (const float*)d_in[10];
    const float* cr4w = (const float*)d_in[11];
    const float* cr4b = (const float*)d_in[12];
    const float* cw5  = (const float*)d_in[13];
    const float* cr5w = (const float*)d_in[14];
    const float* cr5b = (const float*)d_in[15];
    const float* fw1  = (const float*)d_in[16];
    const float* fb1  = (const float*)d_in[17];
    const float* fw2  = (const float*)d_in[18];
    const float* fb2  = (const float*)d_in[19];
    const float* fw3  = (const float*)d_in[20];
    const float* fb3  = (const float*)d_in[21];
    float* out = (float*)d_out;

    // bump allocator over workspace
    char* wp = (char*)d_ws;
    auto alloc = [&](size_t bytes) -> void* {
        void* r = (void*)wp;
        wp += (bytes + 255) & ~(size_t)255;
        return r;
    };
    float* h1 = (float*)alloc((size_t)B * 64 * 256 * 4);
    float* p1 = (float*)alloc((size_t)B * 64 * 64 * 4);
    float* h2 = (float*)alloc((size_t)B * 192 * 64 * 4);
    float* p2 = (float*)alloc((size_t)B * 192 * 16 * 4);
    float* h3 = (float*)alloc((size_t)B * 384 * 16 * 4);
    float* h4 = (float*)alloc((size_t)B * 256 * 16 * 4);
    float* h5 = (float*)alloc((size_t)B * 256 * 16 * 4);
    __bf16* p3  = (__bf16*)alloc((size_t)B * 1024 * 2);
    __bf16* a1  = (__bf16*)alloc((size_t)B * 4096 * 2);
    __bf16* a2  = (__bf16*)alloc((size_t)B * 4096 * 2);
    __bf16* w1b = (__bf16*)alloc((size_t)4096 * 1024 * 2);
    __bf16* w2b = (__bf16*)alloc((size_t)4096 * 4096 * 2);
    __bf16* w3b = (__bf16*)alloc((size_t)100 * 4096 * 2);
    float* r1 = (float*)alloc((size_t)B * 3 * 4);
    float* r2 = (float*)alloc((size_t)B * 3 * 4);
    float* r3 = (float*)alloc((size_t)B * 3 * 4);
    float* r4 = (float*)alloc((size_t)B * 3 * 4);
    float* r5 = (float*)alloc((size_t)B * 3 * 4);

    // FC weight conversion (independent of conv pipeline)
    {
        long n1 = 4096L * 1024, n2 = 4096L * 4096, n3 = 100L * 4096;
        f32_to_bf16_kernel<<<dim3((unsigned)((n1 + 255) / 256)), dim3(256), 0, stream>>>(fw1, w1b, n1);
        f32_to_bf16_kernel<<<dim3((unsigned)((n2 + 255) / 256)), dim3(256), 0, stream>>>(fw2, w2b, n2);
        f32_to_bf16_kernel<<<dim3((unsigned)((n3 + 255) / 256)), dim3(256), 0, stream>>>(fw3, w3b, n3);
    }

    // ---- layer 1: DyRes conv 3->64, s2, p1, 32x32 -> 16x16, ReLU ----
    routing_kernel<<<dim3(B), dim3(256), 0, stream>>>(x, 3, 1024, rw1, rb1, 0, r1);
    mixed_conv_wmma<3, 32, 32, 64, 16, 16, 2, 27, 4, 4>
        <<<dim3(1, 4, B), dim3(128), 0, stream>>>(x, dw1, r1, h1);
    maxpool2_kernel<<<dim3((unsigned)(((long)B * 64 * 64 + 255) / 256)), dim3(256), 0, stream>>>(
        h1, p1, 64, 16, 16, (long)B * 64 * 64);

    // ---- layer 2: DyRes conv 64->192, s1, p1, 8x8 -> 8x8, ReLU ----
    routing_kernel<<<dim3(B), dim3(256), 0, stream>>>(p1, 64, 64, rw2, rb2, 0, r2);
    mixed_conv_wmma<64, 8, 8, 192, 8, 8, 1, 576, 4, 4>
        <<<dim3(3, 1, B), dim3(128), 0, stream>>>(p1, dw2, r2, h2);
    maxpool2_kernel<<<dim3((unsigned)(((long)B * 192 * 16 + 255) / 256)), dim3(256), 0, stream>>>(
        h2, p2, 192, 8, 8, (long)B * 192 * 16);

    // ---- layer 3: DyRes conv 192->384, s1, p1, 4x4 -> 4x4, ReLU ----
    routing_kernel<<<dim3(B), dim3(256), 0, stream>>>(p2, 192, 16, rw3, rb3, 0, r3);
    mixed_conv_wmma<192, 4, 4, 384, 4, 4, 1, 1728, 1, 4>
        <<<dim3(6, 1, B), dim3(128), 0, stream>>>(p2, dw3, r3, h3);

    // ---- layer 4: CondConv 384->256, s1, p1, ReLU ----
    routing_kernel<<<dim3(B), dim3(256), 0, stream>>>(h3, 384, 16, cr4w, cr4b, 1, r4);
    mixed_conv_wmma<384, 4, 4, 256, 4, 4, 1, 3456, 1, 4>
        <<<dim3(4, 1, B), dim3(128), 0, stream>>>(h3, cw4, r4, h4);

    // ---- layer 5: CondConv 256->256, s1, p1, ReLU ----
    routing_kernel<<<dim3(B), dim3(256), 0, stream>>>(h4, 256, 16, cr5w, cr5b, 1, r5);
    mixed_conv_wmma<256, 4, 4, 256, 4, 4, 1, 2304, 1, 4>
        <<<dim3(4, 1, B), dim3(128), 0, stream>>>(h4, cw5, r5, h5);

    // ---- pool + flatten -> bf16 [512,1024] (flatten order == contiguous) ----
    maxpool2_bf16_kernel<<<dim3((unsigned)(((long)B * 256 * 4 + 255) / 256)), dim3(256), 0, stream>>>(
        h5, p3, 256, 4, 4, (long)B * 256 * 4);

    // ---- FC stack ----
    // fc1: [512,1024] x [4096,1024]^T, ReLU -> bf16
    fc_wmma<4, 2><<<dim3(64, 16), dim3(32), 0, stream>>>(p3, w1b, fb1, a1, nullptr, 1024, 4096, 0, 1);
    // fc2: [512,4096] x [4096,4096]^T, ReLU -> bf16
    fc_wmma<4, 2><<<dim3(64, 16), dim3(32), 0, stream>>>(a1, w2b, fb2, a2, nullptr, 4096, 4096, 0, 1);
    // fc3: [512,4096] x [100,4096]^T -> f32 d_out (tiles 0..3 then 4..6)
    fc_wmma<4, 2><<<dim3(1, 16), dim3(32), 0, stream>>>(a2, w3b, fb3, nullptr, out, 4096, 100, 0, 0);
    fc_wmma<3, 2><<<dim3(1, 16), dim3(32), 0, stream>>>(a2, w3b, fb3, nullptr, out, 4096, 100, 4, 0);

    (void)in_sizes; (void)n_in; (void)out_size; (void)ws_size;
}
